// PathAttentionGNN_54202487275746
// MI455X (gfx1250) — matrix operand order repaired and checked by reading, not compiled, not run on hardware
//
#include <hip/hip_runtime.h>
#include <hip/hip_bf16.h>

typedef _Float16 v16h __attribute__((ext_vector_type(16)));
typedef _Float16 v8h  __attribute__((ext_vector_type(8)));
typedef float    v8f  __attribute__((ext_vector_type(8)));

#define NN        50000
#define HIDD      256
#define IND       300
#define KPAD      320
#define G3        768
#define MT_TOTAL  (NN / 16)     // 3125 row-tiles of 16

#define GK   256                // K for all main GEMMs
#define GKS  264                // LDS row stride (+8 halves: 132-word stride -> 4-bank lane offsets, conflict-free)
#define GBN  64                 // block N tile
#define KSP  328                // proj LDS stride (164-word stride -> 36*i bank starts, conflict-free)

static __device__ __forceinline__ v8f wmma16(const v16h& a, const v16h& b, const v8f& c) {
  return __builtin_amdgcn_wmma_f32_16x16x32_f16(false, a, false, b, (short)0, c, false, false);
}

// ---------------------------------------------------------------------------
// WMMA GEMM, K=256: C[M x Nc] = A16[M x 256] * Bt16[Nc x 256]^T + bias[Nc]
// block = 256 threads (8 waves). Block tile 256(M) x 64(N).
// B tile staged once in LDS (contiguous 32KB global read), shared by 8 waves.
// Each wave: 2 M-tiles x 4 N-tiles = 8 WMMA per 32-wide K step.
// ---------------------------------------------------------------------------
__global__ __launch_bounds__(256) void gemm_wmma_lds_kernel(
    const _Float16* __restrict__ A, const _Float16* __restrict__ Bt,
    const float* __restrict__ bias, float* __restrict__ C, int Nc)
{
  __shared__ _Float16 lB[GBN * GKS];
  const int tid   = threadIdx.x;
  const int lane  = tid & 31;
  const int wv    = tid >> 5;
  const int nbase = blockIdx.x * GBN;

  // cooperative, coalesced fill of the 64x256 B tile
  for (int idx = tid; idx < GBN * (GK / 8); idx += 256) {
    const int col = idx >> 5;          // GK/8 == 32 chunks per column
    const int ko  = (idx & 31) * 8;
    v8h v = *(const v8h*)(Bt + (size_t)(nbase + col) * GK + ko);
    *(v8h*)(&lB[col * GKS + ko]) = v;
  }
  __syncthreads();

  const int mt0 = (blockIdx.y * 8 + wv) * 2;
  if (mt0 >= MT_TOTAL) return;
  const bool mt1ok = (mt0 + 1) < MT_TOTAL;

  const int ka   = (lane >> 4) << 3;    // A fragment: k sub-offset 0 / 8
  const int kb   = (lane >> 4) << 3;    // B fragment: k sub-offset 0 / 8
  const int row0 = mt0 * 16 + (lane & 15);
  const int row1 = mt1ok ? row0 + 16 : row0;
  const _Float16* arow0 = A + (size_t)row0 * GK + ka;
  const _Float16* arow1 = A + (size_t)row1 * GK + ka;
  const _Float16* lb0 = &lB[(lane & 15) * GKS + kb];

  v8f a00 = {}, a01 = {}, a02 = {}, a03 = {};
  v8f a10 = {}, a11 = {}, a12 = {}, a13 = {};
#pragma unroll
  for (int k = 0; k < GK; k += 32) {
    v8h x0 = *(const v8h*)(arow0 + k);
    v8h x1 = *(const v8h*)(arow0 + k + 16);
    v8h y0 = *(const v8h*)(arow1 + k);
    v8h y1 = *(const v8h*)(arow1 + k + 16);
    v16h af0, af1;
#pragma unroll
    for (int i = 0; i < 8; ++i) { af0[i] = x0[i]; af0[i + 8] = x1[i]; af1[i] = y0[i]; af1[i + 8] = y1[i]; }
    v16h bf0, bf1, bf2, bf3;
#pragma unroll
    for (int s = 0; s < 4; ++s) {
      v8h blo = *(const v8h*)(lb0 + s * 16 * GKS + k);
      v8h bhi = *(const v8h*)(lb0 + s * 16 * GKS + k + 16);
      v16h* bf = (s == 0) ? &bf0 : (s == 1) ? &bf1 : (s == 2) ? &bf2 : &bf3;
#pragma unroll
      for (int i = 0; i < 8; ++i) { (*bf)[i] = blo[i]; (*bf)[i + 8] = bhi[i]; }
    }
    a00 = wmma16(af0, bf0, a00);
    a01 = wmma16(af0, bf1, a01);
    a02 = wmma16(af0, bf2, a02);
    a03 = wmma16(af0, bf3, a03);
    a10 = wmma16(af1, bf0, a10);
    a11 = wmma16(af1, bf1, a11);
    a12 = wmma16(af1, bf2, a12);
    a13 = wmma16(af1, bf3, a13);
  }

  const int n0 = nbase + (lane & 15);
  const float bb0 = bias[n0], bb1 = bias[n0 + 16], bb2 = bias[n0 + 32], bb3 = bias[n0 + 48];
  const int m0 = mt0 * 16 + ((lane >> 4) << 3);
#pragma unroll
  for (int g = 0; g < 8; ++g) {
    float* crow = C + (size_t)(m0 + g) * Nc + n0;
    crow[0]  = a00[g] + bb0;
    crow[16] = a01[g] + bb1;
    crow[32] = a02[g] + bb2;
    crow[48] = a03[g] + bb3;
  }
  if (mt1ok) {
    const int m1 = m0 + 16;
#pragma unroll
    for (int g = 0; g < 8; ++g) {
      float* crow = C + (size_t)(m1 + g) * Nc + n0;
      crow[0]  = a10[g] + bb0;
      crow[16] = a11[g] + bb1;
      crow[32] = a12[g] + bb2;
      crow[48] = a13[g] + bb3;
    }
  }
}

// ---------------------------------------------------------------------------
// Input projection (K=320): q/e/c = feats @ {Wq,We,Wc}; role-select + ReLU.
// Block = 256 threads; 16-col tile of all three weight matrices staged in LDS.
// Each wave: 2 M-tiles x 3 matrices = 6 WMMA per K step.
// Writes h, path_mem, and their f16 copies (fused cast).
// ---------------------------------------------------------------------------
__global__ __launch_bounds__(256) void proj_select_kernel(
    const _Float16* __restrict__ A,
    const _Float16* __restrict__ Btq, const _Float16* __restrict__ Bte,
    const _Float16* __restrict__ Btc,
    const float* __restrict__ bq, const float* __restrict__ be,
    const float* __restrict__ bc, const int* __restrict__ role,
    float* __restrict__ h, float* __restrict__ pm,
    _Float16* __restrict__ h16, _Float16* __restrict__ pm16)
{
  __shared__ _Float16 lP[3 * 16 * KSP];
  const int tid   = threadIdx.x;
  const int lane  = tid & 31;
  const int wv    = tid >> 5;
  const int nbase = blockIdx.x * 16;

  for (int idx = tid; idx < 3 * 16 * (KPAD / 8); idx += 256) {
    const int mat = idx / (16 * (KPAD / 8));
    const int rem = idx % (16 * (KPAD / 8));
    const int col = rem / (KPAD / 8);
    const int ko  = (rem % (KPAD / 8)) * 8;
    const _Float16* src = (mat == 0) ? Btq : (mat == 1) ? Bte : Btc;
    v8h v = *(const v8h*)(src + (size_t)(nbase + col) * KPAD + ko);
    *(v8h*)(&lP[mat * 16 * KSP + col * KSP + ko]) = v;
  }
  __syncthreads();

  const int mt0 = (blockIdx.y * 8 + wv) * 2;
  if (mt0 >= MT_TOTAL) return;
  const bool mt1ok = (mt0 + 1) < MT_TOTAL;

  const int ka   = (lane >> 4) << 3;
  const int kb   = (lane >> 4) << 3;
  const int row0 = mt0 * 16 + (lane & 15);
  const int row1 = mt1ok ? row0 + 16 : row0;
  const _Float16* arow0 = A + (size_t)row0 * KPAD + ka;
  const _Float16* arow1 = A + (size_t)row1 * KPAD + ka;
  const _Float16* lp0 = &lP[(lane & 15) * KSP + kb];

  v8f q0 = {}, e0 = {}, c0 = {};
  v8f q1 = {}, e1 = {}, c1 = {};
#pragma unroll
  for (int k = 0; k < KPAD; k += 32) {
    v8h x0 = *(const v8h*)(arow0 + k);
    v8h x1 = *(const v8h*)(arow0 + k + 16);
    v8h y0 = *(const v8h*)(arow1 + k);
    v8h y1 = *(const v8h*)(arow1 + k + 16);
    v16h af0, af1;
#pragma unroll
    for (int i = 0; i < 8; ++i) { af0[i] = x0[i]; af0[i + 8] = x1[i]; af1[i] = y0[i]; af1[i + 8] = y1[i]; }
    v16h fq, fe, fc;
#pragma unroll
    for (int s = 0; s < 3; ++s) {
      v8h blo = *(const v8h*)(lp0 + s * 16 * KSP + k);
      v8h bhi = *(const v8h*)(lp0 + s * 16 * KSP + k + 16);
      v16h* bf = (s == 0) ? &fq : (s == 1) ? &fe : &fc;
#pragma unroll
      for (int i = 0; i < 8; ++i) { (*bf)[i] = blo[i]; (*bf)[i + 8] = bhi[i]; }
    }
    q0 = wmma16(af0, fq, q0);
    e0 = wmma16(af0, fe, e0);
    c0 = wmma16(af0, fc, c0);
    q1 = wmma16(af1, fq, q1);
    e1 = wmma16(af1, fe, e1);
    c1 = wmma16(af1, fc, c1);
  }

  const int n0 = nbase + (lane & 15);
  const float bqv = bq[n0], bev = be[n0], bcv = bc[n0];
  const int m0 = mt0 * 16 + ((lane >> 4) << 3);
#pragma unroll
  for (int g = 0; g < 8; ++g) {
    const int m = m0 + g;
    const int rl = role[m];
    float v = (rl == 0) ? (q0[g] + bqv) : ((rl == 1) ? (e0[g] + bev) : (c0[g] + bcv));
    v = fmaxf(v, 0.0f);
    const size_t o = (size_t)m * HIDD + n0;
    h[o] = v; pm[o] = v;
    h16[o] = (_Float16)v; pm16[o] = (_Float16)v;
  }
  if (mt1ok) {
#pragma unroll
    for (int g = 0; g < 8; ++g) {
      const int m = m0 + 16 + g;
      const int rl = role[m];
      float v = (rl == 0) ? (q1[g] + bqv) : ((rl == 1) ? (e1[g] + bev) : (c1[g] + bcv));
      v = fmaxf(v, 0.0f);
      const size_t o = (size_t)m * HIDD + n0;
      h[o] = v; pm[o] = v;
      h16[o] = (_Float16)v; pm16[o] = (_Float16)v;
    }
  }
}

// ---------------------------------------------------------------------------
// Edge scatter: comb[dst] += vt[src] * w  (one wave/edge, hardware f32 atomics,
// vt table and accumulator are both L2-resident)
// ---------------------------------------------------------------------------
__global__ void scatter_kernel(const float* __restrict__ vt,
                               const int* __restrict__ src,
                               const int* __restrict__ dst,
                               const float* __restrict__ w,
                               float* __restrict__ comb, int E)
{
  const int gw   = (int)(((size_t)blockIdx.x * blockDim.x + threadIdx.x) >> 5);
  const int lane = threadIdx.x & 31;
  const int nw   = (int)(((size_t)gridDim.x * blockDim.x) >> 5);
  for (int e = gw; e < E; e += nw) {
    const int s = src[e], d = dst[e];
    const float ww = w[e];
    const float4* vs = (const float4*)(vt + (size_t)s * HIDD);
    float* cd = comb + (size_t)d * HIDD + lane * 8;
    float4 a = vs[lane * 2];
    float4 b = vs[lane * 2 + 1];
    unsafeAtomicAdd(cd + 0, a.x * ww); unsafeAtomicAdd(cd + 1, a.y * ww);
    unsafeAtomicAdd(cd + 2, a.z * ww); unsafeAtomicAdd(cd + 3, a.w * ww);
    unsafeAtomicAdd(cd + 4, b.x * ww); unsafeAtomicAdd(cd + 5, b.y * ww);
    unsafeAtomicAdd(cd + 6, b.z * ww); unsafeAtomicAdd(cd + 7, b.w * ww);
  }
}

// ---------------------------------------------------------------------------
// GRU elementwise: pm = (1-z)*n + z*pm ; also writes pm16 (fused cast)
// ---------------------------------------------------------------------------
__global__ void gru_kernel(const float* __restrict__ gi, const float* __restrict__ gh,
                           float* __restrict__ pm, _Float16* __restrict__ pm16, int n)
{
  const int i = blockIdx.x * blockDim.x + threadIdx.x;
  if (i >= n) return;
  const int row = i >> 8;
  const int j   = i & 255;
  const size_t gb = (size_t)row * G3;
  const float ir = gi[gb + j],        hr = gh[gb + j];
  const float iz = gi[gb + 256 + j],  hz = gh[gb + 256 + j];
  const float in_ = gi[gb + 512 + j], hn = gh[gb + 512 + j];
  const float r = 1.0f / (1.0f + __expf(-(ir + hr)));
  const float z = 1.0f / (1.0f + __expf(-(iz + hz)));
  const float nn = tanhf(in_ + r * hn);
  const float res = (1.0f - z) * nn + z * pm[i];
  pm[i] = res;
  pm16[i] = (_Float16)res;
}

// ---------------------------------------------------------------------------
// LayerNorm: h = LN(h + o) * w + b ; also writes h16 (fused cast).
// On the last layer, also emits out = h_new + path_mem (fused final add).
// ---------------------------------------------------------------------------
__global__ void ln_kernel(float* __restrict__ h, const float* __restrict__ o,
                          const float* __restrict__ w, const float* __restrict__ b,
                          _Float16* __restrict__ h16,
                          const float* __restrict__ pm, float* __restrict__ out)
{
  const int gw   = (blockIdx.x * blockDim.x + threadIdx.x) >> 5;
  const int lane = threadIdx.x & 31;
  if (gw >= NN) return;
  float* hr = h + (size_t)gw * HIDD;
  const float* orow = o + (size_t)gw * HIDD;
  _Float16* h16r = h16 + (size_t)gw * HIDD;
  float x[8];
  float s = 0.0f;
#pragma unroll
  for (int i = 0; i < 8; ++i) { const int j = lane + i * 32; x[i] = hr[j] + orow[j]; s += x[i]; }
#pragma unroll
  for (int off = 16; off > 0; off >>= 1) s += __shfl_xor(s, off, 32);
  const float mu = s * (1.0f / HIDD);
  float v = 0.0f;
#pragma unroll
  for (int i = 0; i < 8; ++i) { const float d = x[i] - mu; v += d * d; }
#pragma unroll
  for (int off = 16; off > 0; off >>= 1) v += __shfl_xor(v, off, 32);
  const float inv = rsqrtf(v * (1.0f / HIDD) + 1e-5f);
  if (out) {
    const float* pmr = pm + (size_t)gw * HIDD;
    float* outr = out + (size_t)gw * HIDD;
#pragma unroll
    for (int i = 0; i < 8; ++i) {
      const int j = lane + i * 32;
      const float r = (x[i] - mu) * inv * w[j] + b[j];
      hr[j] = r;
      h16r[j] = (_Float16)r;
      outr[j] = r + pmr[j];
    }
  } else {
#pragma unroll
    for (int i = 0; i < 8; ++i) {
      const int j = lane + i * 32;
      const float r = (x[i] - mu) * inv * w[j] + b[j];
      hr[j] = r;
      h16r[j] = (_Float16)r;
    }
  }
}

// ---------------------------------------------------------------------------
// Small prep / utility kernels
// ---------------------------------------------------------------------------
__global__ void cast_f16_kernel(const float* __restrict__ x, _Float16* __restrict__ y, int n)
{
  const int i = blockIdx.x * blockDim.x + threadIdx.x;
  if (i < n) y[i] = (_Float16)x[i];
}

__global__ void pad_feats_kernel(const float* __restrict__ x, _Float16* __restrict__ y, int n)
{
  const int i = blockIdx.x * blockDim.x + threadIdx.x;
  if (i >= n) return;
  const int r = i / KPAD, k = i % KPAD;
  y[i] = (k < IND) ? (_Float16)x[(size_t)r * IND + k] : (_Float16)0.0f;
}

// Bt[n*Kp + k] = W[k*Nc + n] (zero-padded past K)
__global__ void transpose_cast_kernel(const float* __restrict__ W, _Float16* __restrict__ Bt,
                                      int K, int Kp, int Nc)
{
  const int i = blockIdx.x * blockDim.x + threadIdx.x;
  if (i >= Nc * Kp) return;
  const int n = i / Kp, k = i % Kp;
  Bt[i] = (k < K) ? (_Float16)W[(size_t)k * Nc + n] : (_Float16)0.0f;
}

// Btv[n][k] = sum_m Wv[hd][k][m] * Wd[m][j]   (hd=n>>6, j=n&63)
__global__ void wcomb_kernel(const float* __restrict__ Wv, const float* __restrict__ Wd,
                             _Float16* __restrict__ Btv)
{
  const int i = blockIdx.x * blockDim.x + threadIdx.x;
  if (i >= HIDD * HIDD) return;
  const int n = i / HIDD, k = i % HIDD;
  const int hd = n >> 6, j = n & 63;
  const float* wv = Wv + ((size_t)hd * HIDD + k) * 64;
  float s = 0.0f;
#pragma unroll 8
  for (int m = 0; m < 64; ++m) s += wv[m] * Wd[m * 64 + j];
  Btv[i] = (_Float16)s;
}

__global__ void bcomb_kernel(const float* __restrict__ bv, const float* __restrict__ Wd,
                             const float* __restrict__ bd, float* __restrict__ bcomb)
{
  const int n = threadIdx.x;      // 256 threads, 1 block
  const int hd = n >> 6, j = n & 63;
  float s = bd[j];
#pragma unroll 8
  for (int m = 0; m < 64; ++m) s += bv[hd * 64 + m] * Wd[m * 64 + j];
  bcomb[n] = s;
}

__global__ void zero_kernel(float* __restrict__ p, int n)
{
  const int i = blockIdx.x * blockDim.x + threadIdx.x;
  if (i < n) p[i] = 0.0f;
}

// ---------------------------------------------------------------------------
extern "C" void kernel_launch(void* const* d_in, const int* in_sizes, int n_in,
                              void* d_out, int out_size, void* d_ws, size_t ws_size,
                              hipStream_t stream)
{
  const float* feats = (const float*)d_in[0];
  const int*   role  = (const int*)d_in[1];
  const int*   esrc  = (const int*)d_in[2];
  const int*   edst  = (const int*)d_in[3];
  const float* ew    = (const float*)d_in[4];
  const float* Wq    = (const float*)d_in[5];
  const float* bq    = (const float*)d_in[6];
  const float* We    = (const float*)d_in[7];
  const float* be    = (const float*)d_in[8];
  const float* Wc    = (const float*)d_in[9];
  const float* bc    = (const float*)d_in[10];
  const float* Wv    = (const float*)d_in[11];
  const float* bv    = (const float*)d_in[12];
  const float* Wd    = (const float*)d_in[13];
  const float* bd    = (const float*)d_in[14];
  const float* Wo    = (const float*)d_in[15];
  const float* bo    = (const float*)d_in[16];
  const float* Wih   = (const float*)d_in[17];
  const float* Whh   = (const float*)d_in[18];
  const float* bih   = (const float*)d_in[19];
  const float* bhh   = (const float*)d_in[20];
  const float* lnw   = (const float*)d_in[21];
  const float* lnb   = (const float*)d_in[22];
  const int E = in_sizes[2];

  // ---- workspace layout ----
  char* ws = (char*)d_ws;
  size_t off = 0;
  auto alloc = [&](size_t bytes) -> char* {
    char* p = ws + off;
    off = (off + bytes + 255) & ~(size_t)255;
    return p;
  };
  float*    h_f   = (float*)alloc((size_t)NN * HIDD * 4);
  float*    pm_f  = (float*)alloc((size_t)NN * HIDD * 4);
  float*    vt_f  = (float*)alloc((size_t)NN * HIDD * 4);   // also reused for 'o'
  float*    cb_f  = (float*)alloc((size_t)NN * HIDD * 4);
  float*    gi_f  = (float*)alloc((size_t)NN * G3 * 4);
  float*    gh_f  = (float*)alloc((size_t)NN * G3 * 4);
  _Float16* a16   = (_Float16*)alloc((size_t)NN * KPAD * 2); // padded feats (proj only)
  _Float16* h16   = (_Float16*)alloc((size_t)NN * HIDD * 2);
  _Float16* c16   = (_Float16*)alloc((size_t)NN * HIDD * 2);
  _Float16* pm16  = (_Float16*)alloc((size_t)NN * HIDD * 2);
  _Float16* btq   = (_Float16*)alloc((size_t)HIDD * KPAD * 2);
  _Float16* bte   = (_Float16*)alloc((size_t)HIDD * KPAD * 2);
  _Float16* btc   = (_Float16*)alloc((size_t)HIDD * KPAD * 2);
  _Float16* btv   = (_Float16*)alloc((size_t)HIDD * HIDD * 2);
  _Float16* wih16 = (_Float16*)alloc((size_t)G3 * HIDD * 2);
  _Float16* whh16 = (_Float16*)alloc((size_t)G3 * HIDD * 2);
  _Float16* wot16 = (_Float16*)alloc((size_t)HIDD * HIDD * 2);
  float*    bcmb  = (float*)alloc((size_t)HIDD * 4);
  (void)ws_size; (void)n_in; (void)out_size;

  const int NH = NN * HIDD;                  // 12.8M
  const dim3 blk1(256);
  const int gy = (MT_TOTAL + 15) / 16;       // 196: 16 M-tiles per block

  // ---- weight prep ----
  {
    const int nfe = NN * KPAD;
    pad_feats_kernel<<<(nfe + 255) / 256, blk1, 0, stream>>>(feats, a16, nfe);
    const int nwp = HIDD * KPAD;
    transpose_cast_kernel<<<(nwp + 255) / 256, blk1, 0, stream>>>(Wq, btq, IND, KPAD, HIDD);
    transpose_cast_kernel<<<(nwp + 255) / 256, blk1, 0, stream>>>(We, bte, IND, KPAD, HIDD);
    transpose_cast_kernel<<<(nwp + 255) / 256, blk1, 0, stream>>>(Wc, btc, IND, KPAD, HIDD);
    wcomb_kernel<<<(HIDD * HIDD + 255) / 256, blk1, 0, stream>>>(Wv, Wd, btv);
    bcomb_kernel<<<1, HIDD, 0, stream>>>(bv, Wd, bd, bcmb);
    const int ng = G3 * HIDD;
    cast_f16_kernel<<<(ng + 255) / 256, blk1, 0, stream>>>(Wih, wih16, ng);
    cast_f16_kernel<<<(ng + 255) / 256, blk1, 0, stream>>>(Whh, whh16, ng);
    transpose_cast_kernel<<<(HIDD * HIDD + 255) / 256, blk1, 0, stream>>>(Wo, wot16, HIDD, HIDD, HIDD);
  }

  // ---- input projection + role select + relu -> h, pm (+f16 copies) ----
  proj_select_kernel<<<dim3(HIDD / 16, gy), blk1, 0, stream>>>(
      a16, btq, bte, btc, bq, be, bc, role, h_f, pm_f, h16, pm16);

  for (int layer = 0; layer < 3; ++layer) {
    // vt = h @ (Wv@Wd concat 4 heads) + (bv@Wd + bd)   [one fused GEMM]
    gemm_wmma_lds_kernel<<<dim3(HIDD / GBN, gy), blk1, 0, stream>>>(
        h16, btv, bcmb, vt_f, HIDD);

    // combined = segment_sum(vt[src] * w, dst)
    zero_kernel<<<(NH + 255) / 256, blk1, 0, stream>>>(cb_f, NH);
    scatter_kernel<<<(E + 7) / 8, blk1, 0, stream>>>(vt_f, esrc, edst, ew, cb_f, E);
    cast_f16_kernel<<<(NH + 255) / 256, blk1, 0, stream>>>(cb_f, c16, NH);

    // GRU: gi = combined @ W_ih^T + b_ih ; gh = path_mem @ W_hh^T + b_hh
    gemm_wmma_lds_kernel<<<dim3(G3 / GBN, gy), blk1, 0, stream>>>(
        c16, wih16, bih, gi_f, G3);
    gemm_wmma_lds_kernel<<<dim3(G3 / GBN, gy), blk1, 0, stream>>>(
        pm16, whh16, bhh, gh_f, G3);
    gru_kernel<<<(NH + 255) / 256, blk1, 0, stream>>>(gi_f, gh_f, pm_f, pm16, NH);

    // h = LN(h + combined @ Wo + bo)   (+ fused f16 cast; layer 2 also emits out)
    gemm_wmma_lds_kernel<<<dim3(HIDD / GBN, gy), blk1, 0, stream>>>(
        c16, wot16, bo, vt_f, HIDD);
    ln_kernel<<<(NN * 32 + 255) / 256, blk1, 0, stream>>>(
        h_f, vt_f, lnw + layer * HIDD, lnb + layer * HIDD, h16,
        pm_f, (layer == 2) ? (float*)d_out : nullptr);
  }
}
